// AtomEncoderLayer_32547262169797
// MI455X (gfx1250) — matrix operand order repaired and checked by reading, not compiled, not run on hardware
//
#include <hip/hip_runtime.h>
#include <hip/hip_bf16.h>

// ---------------- CDNA5 WMMA plumbing (gfx1250, wave32) ----------------
typedef __attribute__((ext_vector_type(16))) __bf16 v16bf;
typedef __attribute__((ext_vector_type(8)))  float  v8f;

#define DEV __device__ __forceinline__

// HW packed f32->bf16 (round-to-nearest-even): one VALU op per 2 elements.
DEV unsigned cvt_pk_bf16(float lo, float hi) {
    unsigned r;
    asm("v_cvt_pk_bf16_f32 %0, %1, %2" : "=v"(r) : "v"(lo), "v"(hi));
    return r;
}
DEV unsigned short f2bf(float f) {
    return (unsigned short)cvt_pk_bf16(f, 0.f);   // low half; ds_store_b16 uses [15:0]
}
DEV float geluf(float x) { return 0.5f * x * (1.0f + erff(x * 0.70710678118654752f)); }
DEV float sigmoidf_(float x) { return 1.0f / (1.0f + __expf(-x)); }

DEV v8f mma_bf16(v16bf a, v16bf b, v8f c) {
    return __builtin_amdgcn_wmma_f32_16x16x32_bf16(false, a, false, b, (short)0, c, false, false);
}

// A fragment (16x32 bf16): lane L -> row M=L&15; half=L>>4; elem i -> K = 8*half + (i<8 ? i : i+8)
DEV v16bf fragA_f32(const float* p, int ld, int lane) {
    union { unsigned u[8]; v16bf v; } f;
    const float* q = p + (lane & 15) * ld + ((lane >> 4) << 3);
#pragma unroll
    for (int j = 0; j < 4; ++j) {
        f.u[j]     = cvt_pk_bf16(q[2 * j],      q[2 * j + 1]);
        f.u[4 + j] = cvt_pk_bf16(q[16 + 2 * j], q[16 + 2 * j + 1]);
    }
    return f.v;
}
DEV v16bf fragA_bf(const unsigned short* p, int ld, int lane) {
    union { unsigned u[8]; v16bf v; } f;
    const unsigned short* q = p + (lane & 15) * ld + ((lane >> 4) << 3);
#pragma unroll
    for (int i = 0; i < 4; ++i) {
        f.u[i]     = *(const unsigned*)(q + 2 * i);
        f.u[4 + i] = *(const unsigned*)(q + 16 + 2 * i);
    }
    return f.v;
}
// B fragment (32x16 bf16, weights stored transposed [OUT][IN]): lane L -> col N=L&15; elem i -> K = 8*half + (i<8 ? i : i+8)
DEV v16bf fragB_bf(const unsigned short* p, int ld, int lane) {
    union { unsigned u[8]; v16bf v; } f;
    const unsigned short* q = p + (lane & 15) * ld + ((lane >> 4) << 3);
#pragma unroll
    for (int i = 0; i < 8; ++i) f.u[i] = *(const unsigned*)(q + 2 * i + (i >= 4 ? 8 : 0));
    return f.v;
}
// C/D: element r of lane L -> M = r + 8*(L>>4), N = L&15

// ---------------- constants ----------------
static constexpr int NB = 4, NN = 512, NK = 32, DIM = 256, PDIM = 128, MSG = 128;
static constexpr int NH = 16, HD = 64, GD = 1024;   // H*HD
static constexpr int BN = NB * NN;                   // 2048

// ---------------- K0: transpose weight [in][out] -> bf16 [out][in] ----------------
__global__ void k_wT(const float* __restrict__ w, unsigned short* __restrict__ wt, int in, int out) {
    int i = blockIdx.x * blockDim.x + threadIdx.x;
    int n = in * out;
    if (i < n) { int r = i / in, c = i - r * in; wt[i] = f2bf(w[(size_t)c * out + r]); }
}

// ---------------- K1/K4a: row-batched projection [BN,256] @ W^T(128x256) -> [BN,128] ----------------
struct ProjArgs {
    const float* A;
    const unsigned short* W0; const unsigned short* W1;
    float* D0; float* D1;
};
__global__ __launch_bounds__(256) void k_proj(ProjArgs p) {
    int lane = threadIdx.x & 31, w = threadIdx.x >> 5;
    int row0 = blockIdx.x * 16;
    const unsigned short* W = blockIdx.y ? p.W1 : p.W0;
    float* D = blockIdx.y ? p.D1 : p.D0;
    const float* Ar = p.A + (size_t)row0 * DIM;
    const unsigned short* Wr = W + (size_t)(w * 16) * DIM;
    v8f acc = {};
#pragma unroll
    for (int ks = 0; ks < DIM / 32; ++ks)
        acc = mma_bf16(fragA_f32(Ar + ks * 32, DIM, lane), fragB_bf(Wr + ks * 32, DIM, lane), acc);
    int n = lane & 15, mh = (lane >> 4) * 8;
#pragma unroll
    for (int r = 0; r < 8; ++r) D[(size_t)(row0 + mh + r) * MSG + w * 16 + n] = acc[r];
}

// ---------------- K2: fused per-node message + GAT kernel ----------------
struct MsgArgs {
    const float* edge; const int* eidx; const int* emask;
    const float* srcp; const float* tgtp;
    const unsigned short *Wemsg, *Wm1, *Wm2, *Wmlin, *Wab, *Wgv;
    const float *ln_g, *ln_b, *b1, *b2;
    float* o_buf;    // [BN,128] raw pooled mean
    float* o2_buf;   // [BN,1024] raw attention output
};
__global__ __launch_bounds__(256) void k_msg(MsgArgs p) {
    __shared__ __attribute__((aligned(16))) float          sM[NK * MSG];       // 16KB
    __shared__ __attribute__((aligned(16))) unsigned short sX[NK * MSG];       // 8KB
    __shared__ __attribute__((aligned(16))) unsigned short sHg[NK * 256];      // 16KB
    __shared__ __attribute__((aligned(16))) unsigned short sMb[NK * MSG];      // 8KB
    __shared__ __attribute__((aligned(16))) float          sAttn[NH * NK];     // 2KB
    __shared__ int sIdx[NK];
    __shared__ int sEm[NK];

    int t = threadIdx.x, lane = t & 31, w = t >> 5;
    int bn = blockIdx.x;
    int b = bn >> 9; // N = 512
    if (t < NK) { sIdx[t] = p.eidx[(size_t)bn * NK + t]; sEm[t] = p.emask[(size_t)bn * NK + t]; }
    __syncthreads();

    const float* E = p.edge + (size_t)bn * NK * PDIM;

    // (A) msg0 = edge @ Wemsg^T + gather(srcp) + tgtp  -> sM (f32)
    {
        v8f a0 = {}, a1 = {};
        const unsigned short* Wr = p.Wemsg + (size_t)(w * 16) * PDIM;
#pragma unroll
        for (int ks = 0; ks < PDIM / 32; ++ks) {
            v16bf fb = fragB_bf(Wr + ks * 32, PDIM, lane);
            a0 = mma_bf16(fragA_f32(E + ks * 32, PDIM, lane), fb, a0);
            a1 = mma_bf16(fragA_f32(E + 16 * PDIM + ks * 32, PDIM, lane), fb, a1);
        }
        int c = w * 16 + (lane & 15), mh = (lane >> 4) * 8;
        const float* tg = p.tgtp + (size_t)bn * MSG;
        float tgc = tg[c];
#pragma unroll
        for (int r = 0; r < 8; ++r) {
            int k0 = mh + r, k1 = 16 + mh + r;
            sM[k0 * MSG + c] = a0[r] + p.srcp[(size_t)(b * NN + sIdx[k0]) * MSG + c] + tgc;
            sM[k1 * MSG + c] = a1[r] + p.srcp[(size_t)(b * NN + sIdx[k1]) * MSG + c] + tgc;
        }
    }
    __syncthreads();

    // (B) LayerNorm rows -> sX (bf16)
    {
        int row = t >> 3, s8 = t & 7;
        const float* rp = sM + row * MSG + s8 * 16;
        float sum = 0.f, sq = 0.f;
#pragma unroll
        for (int i = 0; i < 16; ++i) { float v = rp[i]; sum += v; sq += v * v; }
#pragma unroll
        for (int m = 1; m < 8; m <<= 1) { sum += __shfl_xor(sum, m, 32); sq += __shfl_xor(sq, m, 32); }
        float mu = sum * (1.f / 128.f);
        float rstd = rsqrtf(sq * (1.f / 128.f) - mu * mu + 1e-5f);
        unsigned* op = (unsigned*)(sX + row * MSG + s8 * 16);
#pragma unroll
        for (int i = 0; i < 8; ++i) {
            int c = s8 * 16 + 2 * i;
            op[i] = cvt_pk_bf16((rp[2 * i] - mu) * rstd * p.ln_g[c] + p.ln_b[c],
                                (rp[2 * i + 1] - mu) * rstd * p.ln_g[c + 1] + p.ln_b[c + 1]);
        }
    }
    __syncthreads();

    // (C) H = gelu(Y @ Wm1^T + b1) -> sHg  [32 x 256]
    for (int oc = 0; oc < 2; ++oc) {
        int col0 = (oc * 8 + w) * 16;
        v8f a0 = {}, a1 = {};
        const unsigned short* Wr = p.Wm1 + (size_t)col0 * MSG;
#pragma unroll
        for (int ks = 0; ks < MSG / 32; ++ks) {
            v16bf fb = fragB_bf(Wr + ks * 32, MSG, lane);
            a0 = mma_bf16(fragA_bf(sX + ks * 32, MSG, lane), fb, a0);
            a1 = mma_bf16(fragA_bf(sX + 16 * MSG + ks * 32, MSG, lane), fb, a1);
        }
        int c = col0 + (lane & 15), mh = (lane >> 4) * 8;
        float bb = p.b1[c];
#pragma unroll
        for (int r = 0; r < 8; ++r) {
            sHg[(mh + r) * 256 + c]      = f2bf(geluf(a0[r] + bb));
            sHg[(16 + mh + r) * 256 + c] = f2bf(geluf(a1[r] + bb));
        }
    }
    __syncthreads();

    // (D) msg = H @ Wm2^T + b2, edge-masked -> sM (f32) + sMb (bf16)
    {
        v8f a0 = {}, a1 = {};
        const unsigned short* Wr = p.Wm2 + (size_t)(w * 16) * 256;
#pragma unroll
        for (int ks = 0; ks < 256 / 32; ++ks) {
            v16bf fb = fragB_bf(Wr + ks * 32, 256, lane);
            a0 = mma_bf16(fragA_bf(sHg + ks * 32, 256, lane), fb, a0);
            a1 = mma_bf16(fragA_bf(sHg + 16 * 256 + ks * 32, 256, lane), fb, a1);
        }
        int c = w * 16 + (lane & 15), mh = (lane >> 4) * 8;
        float bb = p.b2[c];
#pragma unroll
        for (int r = 0; r < 8; ++r) {
            int k0 = mh + r, k1 = 16 + mh + r;
            float m0 = sEm[k0] ? (a0[r] + bb) : 0.f;
            float m1 = sEm[k1] ? (a1[r] + bb) : 0.f;
            sM[k0 * MSG + c] = m0; sMb[k0 * MSG + c] = f2bf(m0);
            sM[k1 * MSG + c] = m1; sMb[k1 * MSG + c] = f2bf(m1);
        }
    }
    __syncthreads();

    // (E) pooled mean over valid edges -> o_buf (raw; gated later)
    if (t < MSG) {
        float s = 0.f, cnt = 0.f;
        for (int k = 0; k < NK; ++k) { s += sM[k * MSG + t]; cnt += (float)sEm[k]; }
        p.o_buf[(size_t)bn * MSG + t] = s / (cnt + 1e-6f);
    }

    // (F) T = leaky_relu(msg @ Wmlin^T) -> sX (bf16)
    {
        v8f a0 = {}, a1 = {};
        const unsigned short* Wr = p.Wmlin + (size_t)(w * 16) * MSG;
#pragma unroll
        for (int ks = 0; ks < MSG / 32; ++ks) {
            v16bf fb = fragB_bf(Wr + ks * 32, MSG, lane);
            a0 = mma_bf16(fragA_bf(sMb + ks * 32, MSG, lane), fb, a0);
            a1 = mma_bf16(fragA_bf(sMb + 16 * MSG + ks * 32, MSG, lane), fb, a1);
        }
        int c = w * 16 + (lane & 15), mh = (lane >> 4) * 8;
#pragma unroll
        for (int r = 0; r < 8; ++r) {
            float v0 = a0[r], v1 = a1[r];
            sX[(mh + r) * MSG + c]      = f2bf(v0 > 0.f ? v0 : 0.01f * v0);
            sX[(16 + mh + r) * MSG + c] = f2bf(v1 > 0.f ? v1 : 0.01f * v1);
        }
    }
    __syncthreads();

    // (G) bias = T @ Wab^T [32 x 16], masked -> sAttn[h][k]
    if (w < 2) {
        v8f a = {};
#pragma unroll
        for (int ks = 0; ks < MSG / 32; ++ks)
            a = mma_bf16(fragA_bf(sX + w * 16 * MSG + ks * 32, MSG, lane),
                         fragB_bf(p.Wab + ks * 32, MSG, lane), a);
        int h = lane & 15, mh = (lane >> 4) * 8;
#pragma unroll
        for (int r = 0; r < 8; ++r) {
            int k = w * 16 + mh + r;
            sAttn[h * NK + k] = sEm[k] ? a[r] : -3.0e38f;
        }
    }
    __syncthreads();

    // (H) softmax over k per head
    if (t < NH) {
        float mx = -3.4e38f;
        for (int k = 0; k < NK; ++k) mx = fmaxf(mx, sAttn[t * NK + k]);
        float s = 0.f;
        for (int k = 0; k < NK; ++k) { float e = __expf(sAttn[t * NK + k] - mx); sAttn[t * NK + k] = e; s += e; }
        float inv = 1.f / s;
        for (int k = 0; k < NK; ++k) sAttn[t * NK + k] *= inv;
    }
    __syncthreads();

    // (I) v = msg @ Wgv^T [32 x 1024], fused attn reduce over k -> o2_buf
    for (int oc = 0; oc < 8; ++oc) {
        int col0 = (oc * 8 + w) * 16;
        int head = col0 >> 6;
        const unsigned short* Wr = p.Wgv + (size_t)col0 * MSG;
        float part = 0.f;
#pragma unroll
        for (int rt = 0; rt < 2; ++rt) {
            v8f a = {};
#pragma unroll
            for (int ks = 0; ks < MSG / 32; ++ks)
                a = mma_bf16(fragA_bf(sMb + rt * 16 * MSG + ks * 32, MSG, lane),
                             fragB_bf(Wr + ks * 32, MSG, lane), a);
            int mh = (lane >> 4) * 8;
#pragma unroll
            for (int r = 0; r < 8; ++r) part += a[r] * sAttn[head * NK + rt * 16 + mh + r];
        }
        part += __shfl_xor(part, 16, 32);
        if (lane < 16) p.o2_buf[(size_t)bn * GD + col0 + lane] = part;
    }
}

// ---------------- K3a: gating + dh, node_pre = node + og@Wout^T + o2g@Wgo^T ----------------
struct GateArgs {
    const float* node; const float* o_buf; const float* o2_buf;
    const unsigned short *Wgate, *Wgg, *Wout, *Wgo;
    const float *gb, *ggb;
    float* node_pre;
};
__global__ __launch_bounds__(256) void k_gate(GateArgs p) {
    __shared__ __attribute__((aligned(16))) unsigned short sOg[16 * MSG];  // 4KB
    __shared__ __attribute__((aligned(16))) unsigned short sO2[16 * GD];   // 32KB
    int t = threadIdx.x, lane = t & 31, w = t >> 5;
    int row0 = blockIdx.x * 16;
    const float* A = p.node + (size_t)row0 * DIM;
    {   // og = sigmoid(node@Wgate^T + gb) * o
        v8f a = {};
        const unsigned short* Wr = p.Wgate + (size_t)(w * 16) * DIM;
#pragma unroll
        for (int ks = 0; ks < DIM / 32; ++ks)
            a = mma_bf16(fragA_f32(A + ks * 32, DIM, lane), fragB_bf(Wr + ks * 32, DIM, lane), a);
        int c = w * 16 + (lane & 15), mh = (lane >> 4) * 8;
        float bb = p.gb[c];
#pragma unroll
        for (int r = 0; r < 8; ++r) {
            int m = mh + r;
            sOg[m * MSG + c] = f2bf(sigmoidf_(a[r] + bb) * p.o_buf[(size_t)(row0 + m) * MSG + c]);
        }
    }
    for (int oc = 0; oc < 8; ++oc) {  // o2g = sigmoid(node@Wgg^T + ggb) * o2
        int col0 = (oc * 8 + w) * 16;
        v8f a = {};
        const unsigned short* Wr = p.Wgg + (size_t)col0 * DIM;
#pragma unroll
        for (int ks = 0; ks < DIM / 32; ++ks)
            a = mma_bf16(fragA_f32(A + ks * 32, DIM, lane), fragB_bf(Wr + ks * 32, DIM, lane), a);
        int c = col0 + (lane & 15), mh = (lane >> 4) * 8;
        float bb = p.ggb[c];
#pragma unroll
        for (int r = 0; r < 8; ++r) {
            int m = mh + r;
            sO2[m * GD + c] = f2bf(sigmoidf_(a[r] + bb) * p.o2_buf[(size_t)(row0 + m) * GD + c]);
        }
    }
    __syncthreads();
    for (int oc = 0; oc < 2; ++oc) {  // dh and residual
        int col0 = (oc * 8 + w) * 16;
        v8f a = {};
        const unsigned short* W1r = p.Wout + (size_t)col0 * MSG;
#pragma unroll
        for (int ks = 0; ks < MSG / 32; ++ks)
            a = mma_bf16(fragA_bf(sOg + ks * 32, MSG, lane), fragB_bf(W1r + ks * 32, MSG, lane), a);
        const unsigned short* W2r = p.Wgo + (size_t)col0 * GD;
#pragma unroll
        for (int ks = 0; ks < GD / 32; ++ks)
            a = mma_bf16(fragA_bf(sO2 + ks * 32, GD, lane), fragB_bf(W2r + ks * 32, GD, lane), a);
        int c = col0 + (lane & 15), mh = (lane >> 4) * 8;
#pragma unroll
        for (int r = 0; r < 8; ++r) {
            int m = mh + r;
            p.node_pre[(size_t)(row0 + m) * DIM + c] = p.node[(size_t)(row0 + m) * DIM + c] + a[r];
        }
    }
}

// ---------------- K3b: node MLP + residual + mask -> d_out node ----------------
struct NodeMlpArgs {
    const float* node_pre; const int* mask;
    const unsigned short *W1, *W2;
    const float *ln_g, *ln_b, *b1, *b2;
    float* out_node;
};
__global__ __launch_bounds__(256) void k_nodemlp(NodeMlpArgs p) {
    __shared__ __attribute__((aligned(16))) unsigned short sY[16 * DIM];  // 8KB
    __shared__ __attribute__((aligned(16))) unsigned short sH[16 * 512];  // 16KB
    int t = threadIdx.x, lane = t & 31, w = t >> 5;
    int row0 = blockIdx.x * 16;
    {   // LN over 256, 16 threads/row
        int row = t >> 4, s = t & 15;
        const float* rp = p.node_pre + (size_t)(row0 + row) * DIM + s * 16;
        float sum = 0.f, sq = 0.f;
#pragma unroll
        for (int i = 0; i < 16; ++i) { float v = rp[i]; sum += v; sq += v * v; }
#pragma unroll
        for (int m = 1; m < 16; m <<= 1) { sum += __shfl_xor(sum, m, 32); sq += __shfl_xor(sq, m, 32); }
        float mu = sum * (1.f / 256.f);
        float rstd = rsqrtf(sq * (1.f / 256.f) - mu * mu + 1e-5f);
        unsigned* op = (unsigned*)(sY + row * DIM + s * 16);
#pragma unroll
        for (int i = 0; i < 8; ++i) {
            int c = s * 16 + 2 * i;
            op[i] = cvt_pk_bf16((rp[2 * i] - mu) * rstd * p.ln_g[c] + p.ln_b[c],
                                (rp[2 * i + 1] - mu) * rstd * p.ln_g[c + 1] + p.ln_b[c + 1]);
        }
    }
    __syncthreads();
    for (int oc = 0; oc < 4; ++oc) {  // H = gelu(Y@W1^T + b1), OUT=512
        int col0 = (oc * 8 + w) * 16;
        v8f a = {};
        const unsigned short* Wr = p.W1 + (size_t)col0 * DIM;
#pragma unroll
        for (int ks = 0; ks < DIM / 32; ++ks)
            a = mma_bf16(fragA_bf(sY + ks * 32, DIM, lane), fragB_bf(Wr + ks * 32, DIM, lane), a);
        int c = col0 + (lane & 15), mh = (lane >> 4) * 8;
        float bb = p.b1[c];
#pragma unroll
        for (int r = 0; r < 8; ++r) sH[(mh + r) * 512 + c] = f2bf(geluf(a[r] + bb));
    }
    __syncthreads();
    for (int oc = 0; oc < 2; ++oc) {  // out = node_pre + H@W2^T + b2, masked
        int col0 = (oc * 8 + w) * 16;
        v8f a = {};
        const unsigned short* Wr = p.W2 + (size_t)col0 * 512;
#pragma unroll
        for (int ks = 0; ks < 512 / 32; ++ks)
            a = mma_bf16(fragA_bf(sH + ks * 32, 512, lane), fragB_bf(Wr + ks * 32, 512, lane), a);
        int c = col0 + (lane & 15), mh = (lane >> 4) * 8;
        float bb = p.b2[c];
#pragma unroll
        for (int r = 0; r < 8; ++r) {
            int m = mh + r;
            float v = p.node_pre[(size_t)(row0 + m) * DIM + c] + a[r] + bb;
            p.out_node[(size_t)(row0 + m) * DIM + c] = p.mask[row0 + m] ? v : 0.f;
        }
    }
}

// ---------------- K4b: fused edge update MLP -> d_out edge ----------------
struct EdgeArgs {
    const float* edge; const int* eidx;
    const float* srcp; const float* tgtp;
    const unsigned short *W1, *W2;
    const float *ln_g, *ln_b, *b1, *b2;
    float* out_edge;
};
__global__ __launch_bounds__(256) void k_edge(EdgeArgs p) {
    __shared__ __attribute__((aligned(16))) float          sM[NK * PDIM];  // 16KB
    __shared__ __attribute__((aligned(16))) unsigned short sY[NK * PDIM];  // 8KB
    __shared__ __attribute__((aligned(16))) unsigned short sH[NK * 256];   // 16KB
    __shared__ int sIdx[NK];
    int t = threadIdx.x, lane = t & 31, w = t >> 5;
    int bn = blockIdx.x, b = bn >> 9;
    if (t < NK) sIdx[t] = p.eidx[(size_t)bn * NK + t];
    __syncthreads();
    const float* E = p.edge + (size_t)bn * NK * PDIM;
    {   // emsg = edge + gather(src2) + tgt2
        const float* tg = p.tgtp + (size_t)bn * PDIM;
        for (int e = t; e < NK * PDIM; e += 256) {
            int k = e >> 7, c = e & 127;
            sM[e] = E[e] + p.srcp[(size_t)(b * NN + sIdx[k]) * PDIM + c] + tg[c];
        }
    }
    __syncthreads();
    {   // LN -> sY
        int row = t >> 3, s8 = t & 7;
        const float* rp = sM + row * PDIM + s8 * 16;
        float sum = 0.f, sq = 0.f;
#pragma unroll
        for (int i = 0; i < 16; ++i) { float v = rp[i]; sum += v; sq += v * v; }
#pragma unroll
        for (int m = 1; m < 8; m <<= 1) { sum += __shfl_xor(sum, m, 32); sq += __shfl_xor(sq, m, 32); }
        float mu = sum * (1.f / 128.f);
        float rstd = rsqrtf(sq * (1.f / 128.f) - mu * mu + 1e-5f);
        unsigned* op = (unsigned*)(sY + row * PDIM + s8 * 16);
#pragma unroll
        for (int i = 0; i < 8; ++i) {
            int c = s8 * 16 + 2 * i;
            op[i] = cvt_pk_bf16((rp[2 * i] - mu) * rstd * p.ln_g[c] + p.ln_b[c],
                                (rp[2 * i + 1] - mu) * rstd * p.ln_g[c + 1] + p.ln_b[c + 1]);
        }
    }
    __syncthreads();
    for (int oc = 0; oc < 2; ++oc) {  // H = gelu(Y@W1^T+b1) [32x256]
        int col0 = (oc * 8 + w) * 16;
        v8f a0 = {}, a1 = {};
        const unsigned short* Wr = p.W1 + (size_t)col0 * PDIM;
#pragma unroll
        for (int ks = 0; ks < PDIM / 32; ++ks) {
            v16bf fb = fragB_bf(Wr + ks * 32, PDIM, lane);
            a0 = mma_bf16(fragA_bf(sY + ks * 32, PDIM, lane), fb, a0);
            a1 = mma_bf16(fragA_bf(sY + 16 * PDIM + ks * 32, PDIM, lane), fb, a1);
        }
        int c = col0 + (lane & 15), mh = (lane >> 4) * 8;
        float bb = p.b1[c];
#pragma unroll
        for (int r = 0; r < 8; ++r) {
            sH[(mh + r) * 256 + c]      = f2bf(geluf(a0[r] + bb));
            sH[(16 + mh + r) * 256 + c] = f2bf(geluf(a1[r] + bb));
        }
    }
    __syncthreads();
    {   // edge_out = edge + H@W2^T + b2
        v8f a0 = {}, a1 = {};
        const unsigned short* Wr = p.W2 + (size_t)(w * 16) * 256;
#pragma unroll
        for (int ks = 0; ks < 256 / 32; ++ks) {
            v16bf fb = fragB_bf(Wr + ks * 32, 256, lane);
            a0 = mma_bf16(fragA_bf(sH + ks * 32, 256, lane), fb, a0);
            a1 = mma_bf16(fragA_bf(sH + 16 * 256 + ks * 32, 256, lane), fb, a1);
        }
        int c = w * 16 + (lane & 15), mh = (lane >> 4) * 8;
        float bb = p.b2[c];
        float* O = p.out_edge + (size_t)bn * NK * PDIM;
#pragma unroll
        for (int r = 0; r < 8; ++r) {
            int k0 = mh + r, k1 = 16 + mh + r;
            O[k0 * PDIM + c] = E[k0 * PDIM + c] + a0[r] + bb;
            O[k1 * PDIM + c] = E[k1 * PDIM + c] + a1[r] + bb;
        }
    }
}

// ---------------- host ----------------
extern "C" void kernel_launch(void* const* d_in, const int* in_sizes, int n_in,
                              void* d_out, int out_size, void* d_ws, size_t ws_size,
                              hipStream_t stream) {
    (void)in_sizes; (void)n_in; (void)out_size; (void)ws_size;
    const float* node   = (const float*)d_in[0];
    const float* edge   = (const float*)d_in[1];
    const int*   eidx   = (const int*)d_in[2];
    const int*   emask  = (const int*)d_in[3];
    const int*   mask   = (const int*)d_in[4];
    // params flattened in dict order
    const float* node_src_w = (const float*)d_in[5];
    const float* node_tgt_w = (const float*)d_in[6];
    const float* edge_msg_w = (const float*)d_in[7];
    const float* m_ln_g = (const float*)d_in[8];
    const float* m_ln_b = (const float*)d_in[9];
    const float* m_w1   = (const float*)d_in[10];
    const float* m_b1   = (const float*)d_in[11];
    const float* m_w2   = (const float*)d_in[12];
    const float* m_b2   = (const float*)d_in[13];
    const float* n_ln_g = (const float*)d_in[14];
    const float* n_ln_b = (const float*)d_in[15];
    const float* n_w1   = (const float*)d_in[16];
    const float* n_b1   = (const float*)d_in[17];
    const float* n_w2   = (const float*)d_in[18];
    const float* n_b2   = (const float*)d_in[19];
    const float* gating_w = (const float*)d_in[20];
    const float* gating_b = (const float*)d_in[21];
    const float* out_w    = (const float*)d_in[22];
    const float* msg_lin_w  = (const float*)d_in[23];
    const float* attn_bias_w = (const float*)d_in[24];
    const float* gat_value_w = (const float*)d_in[25];
    const float* gat_gating_w = (const float*)d_in[26];
    const float* gat_gating_b = (const float*)d_in[27];
    const float* gat_out_w = (const float*)d_in[28];
    const float* out_src_w = (const float*)d_in[29];
    const float* out_tgt_w = (const float*)d_in[30];
    const float* e_ln_g = (const float*)d_in[31];
    const float* e_ln_b = (const float*)d_in[32];
    const float* e_w1   = (const float*)d_in[33];
    const float* e_b1   = (const float*)d_in[34];
    const float* e_w2   = (const float*)d_in[35];
    const float* e_b2   = (const float*)d_in[36];

    char* wsp = (char*)d_ws;
    size_t off = 0;
    auto allocF = [&](size_t n) -> float* { float* p = (float*)(wsp + off); off += n * sizeof(float); return p; };
    auto allocU = [&](size_t n) -> unsigned short* { unsigned short* p = (unsigned short*)(wsp + off); off += n * sizeof(unsigned short); return p; };

    float* srcp     = allocF((size_t)BN * MSG);
    float* tgtp     = allocF((size_t)BN * MSG);
    float* o_buf    = allocF((size_t)BN * MSG);
    float* o2_buf   = allocF((size_t)BN * GD);
    float* node_pre = allocF((size_t)BN * DIM);

    unsigned short* Wsrc  = allocU(128 * 256);
    unsigned short* Wtgt  = allocU(128 * 256);
    unsigned short* Wemsg = allocU(128 * 128);
    unsigned short* Wm1   = allocU(256 * 128);
    unsigned short* Wm2   = allocU(128 * 256);
    unsigned short* Wn1   = allocU(512 * 256);
    unsigned short* Wn2   = allocU(256 * 512);
    unsigned short* Wgate = allocU(128 * 256);
    unsigned short* Wout  = allocU(256 * 128);
    unsigned short* Wmlin = allocU(128 * 128);
    unsigned short* Wab   = allocU(16 * 128);
    unsigned short* Wgv   = allocU(1024 * 128);
    unsigned short* Wgg   = allocU(1024 * 256);
    unsigned short* Wgo   = allocU(256 * 1024);
    unsigned short* Wosrc = allocU(128 * 256);
    unsigned short* Wotgt = allocU(128 * 256);
    unsigned short* We1   = allocU(256 * 128);
    unsigned short* We2   = allocU(128 * 256);

    auto runT = [&](const float* w, unsigned short* wt, int in, int out) {
        int n = in * out;
        k_wT<<<dim3((n + 255) / 256), dim3(256), 0, stream>>>(w, wt, in, out);
    };
    runT(node_src_w, Wsrc, 256, 128);   runT(node_tgt_w, Wtgt, 256, 128);
    runT(edge_msg_w, Wemsg, 128, 128);
    runT(m_w1, Wm1, 128, 256);          runT(m_w2, Wm2, 256, 128);
    runT(n_w1, Wn1, 256, 512);          runT(n_w2, Wn2, 512, 256);
    runT(gating_w, Wgate, 256, 128);    runT(out_w, Wout, 128, 256);
    runT(msg_lin_w, Wmlin, 128, 128);   runT(attn_bias_w, Wab, 128, 16);
    runT(gat_value_w, Wgv, 128, 1024);  runT(gat_gating_w, Wgg, 256, 1024);
    runT(gat_out_w, Wgo, 1024, 256);
    runT(out_src_w, Wosrc, 256, 128);   runT(out_tgt_w, Wotgt, 256, 128);
    runT(e_w1, We1, 128, 256);          runT(e_w2, We2, 256, 128);

    // K1: src/tgt projections of node_repr
    ProjArgs pa; pa.A = node; pa.W0 = Wsrc; pa.W1 = Wtgt; pa.D0 = srcp; pa.D1 = tgtp;
    k_proj<<<dim3(BN / 16, 2), dim3(256), 0, stream>>>(pa);

    // K2: fused per-node message + GAT
    MsgArgs ma;
    ma.edge = edge; ma.eidx = eidx; ma.emask = emask; ma.srcp = srcp; ma.tgtp = tgtp;
    ma.Wemsg = Wemsg; ma.Wm1 = Wm1; ma.Wm2 = Wm2; ma.Wmlin = Wmlin; ma.Wab = Wab; ma.Wgv = Wgv;
    ma.ln_g = m_ln_g; ma.ln_b = m_ln_b; ma.b1 = m_b1; ma.b2 = m_b2;
    ma.o_buf = o_buf; ma.o2_buf = o2_buf;
    k_msg<<<dim3(BN), dim3(256), 0, stream>>>(ma);

    // K3a: gating + dh + residual
    GateArgs ga;
    ga.node = node; ga.o_buf = o_buf; ga.o2_buf = o2_buf;
    ga.Wgate = Wgate; ga.Wgg = Wgg; ga.Wout = Wout; ga.Wgo = Wgo;
    ga.gb = gating_b; ga.ggb = gat_gating_b; ga.node_pre = node_pre;
    k_gate<<<dim3(BN / 16), dim3(256), 0, stream>>>(ga);

    // K3b: node MLP + mask -> d_out node section
    float* out_node = (float*)d_out;
    float* out_edge = out_node + (size_t)BN * DIM;
    NodeMlpArgs na;
    na.node_pre = node_pre; na.mask = mask; na.W1 = Wn1; na.W2 = Wn2;
    na.ln_g = n_ln_g; na.ln_b = n_ln_b; na.b1 = n_b1; na.b2 = n_b2; na.out_node = out_node;
    k_nodemlp<<<dim3(BN / 16), dim3(256), 0, stream>>>(na);

    // K4a: projections of updated nodes (reuse srcp/tgtp buffers)
    ProjArgs pb; pb.A = out_node; pb.W0 = Wosrc; pb.W1 = Wotgt; pb.D0 = srcp; pb.D1 = tgtp;
    k_proj<<<dim3(BN / 16, 2), dim3(256), 0, stream>>>(pb);

    // K4b: fused edge update -> d_out edge section
    EdgeArgs ea;
    ea.edge = edge; ea.eidx = eidx; ea.srcp = srcp; ea.tgtp = tgtp;
    ea.W1 = We1; ea.W2 = We2;
    ea.ln_g = e_ln_g; ea.ln_b = e_ln_b; ea.b1 = e_b1; ea.b2 = e_b2;
    ea.out_edge = out_edge;
    k_edge<<<dim3(BN), dim3(256), 0, stream>>>(ea);
}
